// BatchProgramEncoder_10153302688334
// MI455X (gfx1250) — matrix operand order repaired and checked by reading, not compiled, not run on hardware
//
#include <hip/hip_runtime.h>
#include <hip/hip_bf16.h>
#include <math.h>

typedef __attribute__((ext_vector_type(16))) _Float16 v16h;
typedef __attribute__((ext_vector_type(8)))  float    v8f;

// Problem constants (from reference)
#define Bsz 32
#define Lsz 128
#define Tsz 63      // 2^6 - 1 tree nodes
#define Esz 128
#define Csz 128
#define Hsz 256
#define G3H 768     // 3*H

// ---------------------------------------------------------------------------
// Fragment-major LDS layouts: each lane's 16 f16 fragment elements are stored
// contiguously (32 bytes, 32B-aligned) so a fragment load is one v16h LDS read
// (2x ds_load_b128), instead of 16 scalar ds_load_u16 + v_mov_b16 packing.
//
// A 16x32 f16 (M x K): lane = 16*half + M(0..15)
//   element e<8  -> K = 8*half + e
//   element e>=8 -> K = 16 + 8*half + (e-8)
// B 32x16 f16 (K x N): lane = 16*half + N(0..15); element e -> same K as A (swizzled)
// D 16x16 f32: VGPR v -> M = v + 8*half, N = lane&15
// ---------------------------------------------------------------------------

// A-swizzle: given K-within-32 (kin), return (half, e)
__device__ __forceinline__ void a_swz(int kin, int& half, int& e) {
    int group = kin >> 3;               // 0..3
    half = group & 1;
    e    = (kin & 7) + 8 * (group >> 1);
}

// ===========================================================================
// Kernel 1: fused tree encoder.  One block per (b,l).
//   h[t][c] = sum_e emb[tok[t]][e] * Wc[c][e] + bc[c]   via WMMA (f16 in, f32 acc)
//   binary-tree child->parent accumulate, max over 63 nodes, ReLU -> enc[b,l,c]
// ===========================================================================
__global__ __launch_bounds__(256)
void tree_encode_kernel(const int* __restrict__ tokens,
                        const float* __restrict__ emb,
                        const float* __restrict__ Wc,
                        const float* __restrict__ bc,
                        float* __restrict__ enc)
{
    // sAf: fragment-major A, [mt(4)][kb(4)][lane(32)][e(16)] = 8192 halves, 16 KB
    __shared__ __align__(32) _Float16 sAf[4 * 4 * 32 * 16];
    // sBH: first fragment-major B [kb(4)][nt(8)][lane(32)][e(16)] f16 (32 KB),
    //      then reused (after barrier) as f32 H [64][128] (32 KB)
    __shared__ __align__(32) float sBH[64 * 128];
    __shared__ int s_tok[64];
    _Float16* sBf = (_Float16*)sBH;
    float*    sH  = (float*)sBH;

    const int bl   = blockIdx.x;             // b*L + l
    const int tid  = threadIdx.x;
    const int lane = tid & 31;
    const int wave = tid >> 5;
    const int r    = lane & 15;
    const int half = lane >> 4;

    if (tid < 64) s_tok[tid] = (tid < Tsz) ? tokens[bl * Tsz + tid] : 0;
    __syncthreads();

    // Stage A (embedding gather, L2-resident table) -> fragment-major f16
    for (int idx = tid; idx < 64 * 128; idx += 256) {
        int row = idx >> 7, col = idx & 127;          // M=row, K=col
        float v = (row < Tsz) ? emb[(long)s_tok[row] * Esz + col] : 0.0f;
        int mt = row >> 4, rr = row & 15;
        int kb = col >> 5, kin = col & 31;
        int h2, e;  a_swz(kin, h2, e);
        sAf[(((mt * 4 + kb) * 32) + (h2 * 16 + rr)) * 16 + e] = (_Float16)v;
    }
    // Stage B: B[k][n] = Wc[n][k] -> fragment-major f16
    for (int idx = tid; idx < 128 * 128; idx += 256) {
        int c = idx >> 7, k = idx & 127;              // N=c, K=k (coalesced read)
        int nt = c >> 4, rr = c & 15;
        int kb = k >> 5, kin = k & 31;
        int h2, e;  a_swz(kin, h2, e);
        sBf[(((kb * 8 + nt) * 32) + (h2 * 16 + rr)) * 16 + e] = (_Float16)Wc[idx];
    }
    __syncthreads();

    // 4 mt x 8 nt = 32 output tiles, 4 per wave; K = 128 = 4 x 32
    v8f acc[4];
#pragma unroll
    for (int i = 0; i < 4; ++i) {
        const int tile = wave + i * 8;
        const int mt = tile >> 3, nt = tile & 7;
        v8f c = {};
#pragma unroll
        for (int kb = 0; kb < 4; ++kb) {
            const v16h a = *(const v16h*)(sAf + (((mt * 4 + kb) * 32 + lane) << 4));
            const v16h b = *(const v16h*)(sBf + (((kb * 8 + nt) * 32 + lane) << 4));
            c = __builtin_amdgcn_wmma_f32_16x16x32_f16(false, a, false, b,
                                                       (short)0, c, false, false);
        }
        acc[i] = c;
    }
    __syncthreads();  // all sBf reads done before reusing the space as sH

#pragma unroll
    for (int i = 0; i < 4; ++i) {
        const int tile = wave + i * 8;
        const int mt = tile >> 3, nt = tile & 7;
        const float bcv = bc[nt * 16 + r];
#pragma unroll
        for (int v = 0; v < 8; ++v) {
            int mrow = mt * 16 + 8 * half + v;
            sH[mrow * 128 + nt * 16 + r] = acc[i][v] + bcv;
        }
    }
    __syncthreads();

    // In-LDS binary-tree accumulate + max + ReLU; one thread per channel
    if (tid < Csz) {
        const int c = tid;
        for (int d = 5; d >= 1; --d) {
            int start = (1 << d) - 1;
            int p0    = (1 << (d - 1)) - 1;
            for (int i = 0; i < (1 << (d - 1)); ++i) {
                sH[(p0 + i) * 128 + c] +=
                    sH[(start + 2 * i) * 128 + c] + sH[(start + 2 * i + 1) * 128 + c];
            }
        }
        float m = -3.4e38f;
        for (int node = 0; node < Tsz; ++node) m = fmaxf(m, sH[node * 128 + c]);
        enc[bl * Csz + c] = fmaxf(m, 0.0f);
    }
}

// ===========================================================================
// Kernel 2: GRU input projections (both directions), gi = enc @ Wih^T + bih
//   (4096 x 128) x (128 x 768) via WMMA.  grid = (64 m-blocks, 6 n-blocks, 2)
// ===========================================================================
__global__ __launch_bounds__(256)
void gi_gemm_kernel(const float* __restrict__ enc,
                    const float* __restrict__ Wih_f, const float* __restrict__ bih_f,
                    const float* __restrict__ Wih_b, const float* __restrict__ bih_b,
                    float* __restrict__ gi_f, float* __restrict__ gi_b)
{
    __shared__ __align__(32) _Float16 sAf[4 * 4 * 32 * 16];   // 16 KB
    __shared__ __align__(32) _Float16 sBf[4 * 8 * 32 * 16];   // 32 KB

    const int tid  = threadIdx.x;
    const int lane = tid & 31;
    const int wave = tid >> 5;
    const int r    = lane & 15;
    const int half = lane >> 4;
    const int m0 = blockIdx.x * 64;
    const int j0 = blockIdx.y * 128;
    const float* Wih = blockIdx.z ? Wih_b : Wih_f;
    const float* bih = blockIdx.z ? bih_b : bih_f;
    float*       gi  = blockIdx.z ? gi_b  : gi_f;

    // Stage A: enc rows -> fragment-major
    for (int idx = tid; idx < 64 * 128; idx += 256) {
        int row = idx >> 7, col = idx & 127;
        float v = enc[(m0 + row) * Csz + col];
        int mt = row >> 4, rr = row & 15;
        int kb = col >> 5, kin = col & 31;
        int h2, e;  a_swz(kin, h2, e);
        sAf[(((mt * 4 + kb) * 32) + (h2 * 16 + rr)) * 16 + e] = (_Float16)v;
    }
    // Stage B: B[k][jj] = Wih[j0+jj][k] -> fragment-major
    for (int idx = tid; idx < 128 * 128; idx += 256) {
        int jj = idx >> 7, k = idx & 127;             // coalesced along k
        int nt = jj >> 4, rr = jj & 15;
        int kb = k >> 5, kin = k & 31;
        int h2, e;  a_swz(kin, h2, e);
        sBf[(((kb * 8 + nt) * 32) + (h2 * 16 + rr)) * 16 + e] =
            (_Float16)Wih[(j0 + jj) * Csz + k];
    }
    __syncthreads();

#pragma unroll
    for (int i = 0; i < 4; ++i) {
        const int tile = wave + i * 8;
        const int mt = tile >> 3, nt = tile & 7;
        v8f c = {};
#pragma unroll
        for (int kb = 0; kb < 4; ++kb) {
            const v16h a = *(const v16h*)(sAf + (((mt * 4 + kb) * 32 + lane) << 4));
            const v16h b = *(const v16h*)(sBf + (((kb * 8 + nt) * 32 + lane) << 4));
            c = __builtin_amdgcn_wmma_f32_16x16x32_f16(false, a, false, b,
                                                       (short)0, c, false, false);
        }
        const float bv = bih[j0 + nt * 16 + r];
#pragma unroll
        for (int v = 0; v < 8; ++v) {
            int m = m0 + mt * 16 + 8 * half + v;
            gi[(long)m * G3H + j0 + nt * 16 + r] = c[v] + bv;
        }
    }
}

// ===========================================================================
// Kernel 3a: transpose Whh (768x256 -> 256x768) for coalesced recurrence reads
// ===========================================================================
__global__ void whh_transpose_kernel(const float* __restrict__ Whh_f,
                                     const float* __restrict__ Whh_b,
                                     float* __restrict__ T_f,
                                     float* __restrict__ T_b)
{
    int idx = blockIdx.x * 256 + threadIdx.x;
    if (idx < G3H * Hsz) {
        int j = idx / Hsz, k = idx % Hsz;
        T_f[k * G3H + j] = Whh_f[idx];
        T_b[k * G3H + j] = Whh_b[idx];
    }
}

// ===========================================================================
// Kernel 3b: GRU recurrence + fused max over L.  grid = (B, 2 dirs), 256 thr.
//   hp lives in LDS; thread t owns hidden channel t and its 3 gate dots.
//   Running max replaces the (B,L,512) intermediate entirely.
// ===========================================================================
__global__ __launch_bounds__(256)
void gru_kernel(const float* __restrict__ gi_f, const float* __restrict__ gi_b,
                const float* __restrict__ whhT_f, const float* __restrict__ whhT_b,
                const float* __restrict__ bhh_f, const float* __restrict__ bhh_b,
                float* __restrict__ out)
{
    __shared__ float hp[Hsz];
    const int t   = threadIdx.x;
    const int b   = blockIdx.x;
    const int dir = blockIdx.y;
    const float* gi  = dir ? gi_b   : gi_f;
    const float* WT  = dir ? whhT_b : whhT_f;
    const float* bhh = dir ? bhh_b  : bhh_f;

    hp[t] = 0.0f;
    float runmax = -3.4e38f;
    const float b0 = bhh[t], b1 = bhh[Hsz + t], b2 = bhh[2 * Hsz + t];

    for (int s = 0; s < Lsz; ++s) {
        const int l = dir ? (Lsz - 1 - s) : s;
        __syncthreads();  // hp ready for this step
        float g0 = b0, g1 = b1, g2 = b2;
        for (int k = 0; k < Hsz; ++k) {
            const float hk = hp[k];
            const float* w = WT + k * G3H;   // consecutive t -> consecutive addr
            g0 = fmaf(hk, w[t],           g0);
            g1 = fmaf(hk, w[Hsz + t],     g1);
            g2 = fmaf(hk, w[2 * Hsz + t], g2);
        }
        const float* gr = gi + ((long)b * Lsz + l) * G3H;
        const float rr = 1.0f / (1.0f + expf(-(gr[t]           + g0)));
        const float zz = 1.0f / (1.0f + expf(-(gr[Hsz + t]     + g1)));
        const float nn = tanhf(gr[2 * Hsz + t] + rr * g2);
        const float hprev = hp[t];
        const float hnew  = (1.0f - zz) * nn + zz * hprev;
        runmax = fmaxf(runmax, hnew);
        __syncthreads();  // all reads of hp done
        hp[t] = hnew;
    }
    out[b * (2 * Hsz) + dir * Hsz + t] = runmax;
}

// ===========================================================================
extern "C" void kernel_launch(void* const* d_in, const int* in_sizes, int n_in,
                              void* d_out, int out_size, void* d_ws, size_t ws_size,
                              hipStream_t stream)
{
    const int*   tokens = (const int*)  d_in[0];
    const float* emb    = (const float*)d_in[1];
    const float* Wc     = (const float*)d_in[2];
    const float* bc     = (const float*)d_in[3];
    const float* Wih_f  = (const float*)d_in[4];
    const float* Whh_f  = (const float*)d_in[5];
    const float* bih_f  = (const float*)d_in[6];
    const float* bhh_f  = (const float*)d_in[7];
    const float* Wih_b  = (const float*)d_in[8];
    const float* Whh_b  = (const float*)d_in[9];
    const float* bih_b  = (const float*)d_in[10];
    const float* bhh_b  = (const float*)d_in[11];
    float* out = (float*)d_out;

    // Workspace carve-up (floats): ~27.5 MB total
    float* ws    = (float*)d_ws;
    float* enc   = ws;                               // 4096*128
    float* gi_f  = enc  + (size_t)Bsz * Lsz * Csz;   // 4096*768
    float* gi_b  = gi_f + (size_t)Bsz * Lsz * G3H;   // 4096*768
    float* wT_f  = gi_b + (size_t)Bsz * Lsz * G3H;   // 256*768
    float* wT_b  = wT_f + (size_t)Hsz * G3H;

    // 1) fused tree encoder (WMMA GEMM + in-LDS tree reduce)
    tree_encode_kernel<<<Bsz * Lsz, 256, 0, stream>>>(tokens, emb, Wc, bc, enc);

    // 2) batched GRU input projections, both directions (WMMA GEMM)
    gi_gemm_kernel<<<dim3((Bsz * Lsz) / 64, G3H / 128, 2), 256, 0, stream>>>(
        enc, Wih_f, bih_f, Wih_b, bih_b, gi_f, gi_b);

    // 3a) transpose recurrent weights for coalesced access
    whh_transpose_kernel<<<(G3H * Hsz + 255) / 256, 256, 0, stream>>>(
        Whh_f, Whh_b, wT_f, wT_b);

    // 3b) recurrence with fused running-max -> (32, 512) output
    gru_kernel<<<dim3(Bsz, 2), 256, 0, stream>>>(
        gi_f, gi_b, wT_f, wT_b, bhh_f, bhh_b, out);
}
